// CATA_30339648979575
// MI455X (gfx1250) — compile-verified, matrix operand order checked by reading, not run on hardware
//
#include <hip/hip_runtime.h>
#include <cstddef>
#include <cstdint>

// ---------------------------------------------------------------------------
// CDNA5 (gfx1250) wave32 WMMA helpers
// ---------------------------------------------------------------------------
typedef __attribute__((ext_vector_type(16))) _Float16 v16h;
typedef __attribute__((ext_vector_type(8)))  _Float16 v8h;
typedef __attribute__((ext_vector_type(4)))  _Float16 v4h;
typedef __attribute__((ext_vector_type(8)))  float    v8f;

__device__ __forceinline__ v8f wmma_f32_16x16x32(const v16h a, const v16h b, const v8f c) {
    // (neg_a, A, neg_b, B, c_mod, C, reuse_a, reuse_b)
    return __builtin_amdgcn_wmma_f32_16x16x32_f16(false, a, false, b, (short)0, c, false, false);
}

// Load one 16x32 f16 fragment (A layout; identical for Bt stored [n][k]) from
// LDS. Per ISA 7.12.2: lanes 0-15 hold row M=lane with K 0..7 & 16..23,
// lanes 16-31 hold row M=lane-16 with K 8..15 & 24..31. Two ds_load_b128 each.
__device__ __forceinline__ v16h load_frag(const _Float16* base, int ld) {
    const int lane = threadIdx.x & 31;
    const int r  = lane & 15;
    const int kb = (lane >> 4) * 8;
    const _Float16* p = base + (size_t)r * ld;
    v8h lo = *(const v8h*)(p + kb);
    v8h hi = *(const v8h*)(p + kb + 16);
    return __builtin_shufflevector(lo, hi, 0,1,2,3,4,5,6,7,8,9,10,11,12,13,14,15);
}

__device__ __forceinline__ v8h cvt2x4_f16(float4 a, float4 b) {
    v8h h;
    h[0] = (_Float16)a.x; h[1] = (_Float16)a.y; h[2] = (_Float16)a.z; h[3] = (_Float16)a.w;
    h[4] = (_Float16)b.x; h[5] = (_Float16)b.y; h[6] = (_Float16)b.z; h[7] = (_Float16)b.w;
    return h;
}

// Epilogue helpers: constant row stride -> global_store immediate offsets.
template <int STRIDE>
__device__ __forceinline__ void store8(float* p, const v8f& a) {
#pragma unroll
    for (int r = 0; r < 8; ++r) p[(size_t)r * STRIDE] = a[r];
}
template <int STRIDE>
__device__ __forceinline__ void accum8(float* p, const v8f& a) {
#pragma unroll
    for (int r = 0; r < 8; ++r) p[(size_t)r * STRIDE] += a[r];
}

// ---------------------------------------------------------------------------
// Kernel 1: Out_f16[M,N] = X_f32[M,512] * W_f32[N,512]^T + bias (1x1 conv GEMM)
// Block tile 64x64, BK=32, 8 waves (4 M-slabs x 2 N-slabs, 2 WMMAs each).
// ---------------------------------------------------------------------------
template <int N>
__global__ __launch_bounds__(256) void gemm512(
    const float* __restrict__ X, const float* __restrict__ W,
    const float* __restrict__ bias, _Float16* __restrict__ Out)
{
    constexpr int K = 512;
    __shared__ __align__(16) _Float16 As[64][40];
    __shared__ __align__(16) _Float16 Bs[64][40];
    const int tid  = threadIdx.x;
    const int lane = tid & 31;
    const int wave = tid >> 5;
    const int wm = wave >> 1;     // 0..3 : 16-row slab
    const int wn = wave & 1;      // 0..1 : 32-col slab
    const int mBase = blockIdx.x * 64;
    const int nBase = blockIdx.y * 64;

    // staging coordinates: each thread owns 8 consecutive k of one row
    const int sr = tid >> 2;            // 0..63
    const int sc = (tid & 3) * 8;       // 0,8,16,24
    const float* Xrow = X + (size_t)(mBase + sr) * K + sc;
    const float* Wrow = W + (size_t)(nBase + sr) * K + sc;

    v8f acc0 = {}; v8f acc1 = {};

    for (int k0 = 0; k0 < K; k0 += 32) {
        // load phase (all vmem in flight), then convert + LDS store phase
        float4 a0 = *(const float4*)(Xrow + k0);
        float4 a1 = *(const float4*)(Xrow + k0 + 4);
        float4 b0 = *(const float4*)(Wrow + k0);
        float4 b1 = *(const float4*)(Wrow + k0 + 4);
        if (k0 + 32 < K)  // warm next k-slice (global_prefetch_b8)
            __builtin_prefetch(Xrow + k0 + 32, 0, 0);
        v8h ha = cvt2x4_f16(a0, a1);
        v8h hb = cvt2x4_f16(b0, b1);
        *(v8h*)&As[sr][sc] = ha;
        *(v8h*)&Bs[sr][sc] = hb;
        __syncthreads();
        v16h a  = load_frag(&As[wm * 16][0], 40);
        v16h f0 = load_frag(&Bs[wn * 32][0], 40);
        v16h f1 = load_frag(&Bs[wn * 32 + 16][0], 40);
        acc0 = wmma_f32_16x16x32(a, f0, acc0);
        acc1 = wmma_f32_16x16x32(a, f1, acc1);
        __syncthreads();
    }

    const int n0 = lane & 15;
    const int mb = (lane >> 4) * 8;
    const int gm = mBase + wm * 16 + mb;
    const int gn = nBase + wn * 32;
    const float bias0 = bias[gn + n0];
    const float bias1 = bias[gn + 16 + n0];
    _Float16* p0 = Out + (size_t)gm * N + gn + n0;
#pragma unroll
    for (int r = 0; r < 8; ++r) {
        p0[(size_t)r * N]      = (_Float16)(acc0[r] + bias0);
        p0[(size_t)r * N + 16] = (_Float16)(acc1[r] + bias1);
    }
}

// ---------------------------------------------------------------------------
// Kernel 2: criss-cross energies (MODE is a template arg: one specialized
// epilogue per instantiation -> no runtime branch, no SGPR spill pressure).
//  MODE==0 : fixed column j=idx -> eH[b,i,j,k]  (k = row index)
//  MODE==1 : fixed row    i=idx -> eW[b,i,j,k]  (k = col index)
//  E layout: [b][i][j][m], m = k (H part) or 128+k (W part). 128x128x64 GEMM.
// ---------------------------------------------------------------------------
template <int MODE>
__global__ __launch_bounds__(256) void cc_energy(
    const _Float16* __restrict__ Q, const _Float16* __restrict__ Kt,
    float* __restrict__ E)
{
    __shared__ __align__(16) _Float16 Qs[128][72];
    __shared__ __align__(16) _Float16 Ks[128][72];
    const int tid  = threadIdx.x;
    const int b    = blockIdx.x >> 7;
    const int idx  = blockIdx.x & 127;
    const size_t bOff = (size_t)b * 16384;

    // pure f16 copy staging: batch 8 uint4 loads, then 8 LDS stores
    {
        const int r = tid >> 1;               // 0..127
        const int c = (tid & 1) * 32;         // 0 or 32
        const int n = (MODE == 0) ? (r * 128 + idx) : (idx * 128 + r);
        const uint4* qs = (const uint4*)(Q  + (bOff + n) * 64 + c);
        const uint4* ks = (const uint4*)(Kt + (bOff + n) * 64 + c);
        uint4 qb[4], kb4[4];
#pragma unroll
        for (int t = 0; t < 4; ++t) qb[t] = qs[t];
#pragma unroll
        for (int t = 0; t < 4; ++t) kb4[t] = ks[t];
        uint4* qd = (uint4*)&Qs[r][c];
        uint4* kd = (uint4*)&Ks[r][c];
#pragma unroll
        for (int t = 0; t < 4; ++t) qd[t] = qb[t];
#pragma unroll
        for (int t = 0; t < 4; ++t) kd[t] = kb4[t];
    }
    __syncthreads();

    const int lane = tid & 31, wave = tid >> 5;
    const int wm = wave >> 1;   // 0..3 : 32 output rows
    const int wn = wave & 1;    // 0..1 : 64 output cols
    v8f acc[2][4] = {};

#pragma unroll
    for (int k0 = 0; k0 < 64; k0 += 32) {
        v16h af0 = load_frag(&Qs[wm * 32][k0], 72);
        v16h af1 = load_frag(&Qs[wm * 32 + 16][k0], 72);
#pragma unroll
        for (int sn = 0; sn < 4; ++sn) {
            v16h bf = load_frag(&Ks[wn * 64 + sn * 16][k0], 72);
            acc[0][sn] = wmma_f32_16x16x32(af0, bf, acc[0][sn]);
            acc[1][sn] = wmma_f32_16x16x32(af1, bf, acc[1][sn]);
        }
    }

    // epilogue: base pointer + constant per-row stride (immediate offsets)
    const int n0 = lane & 15, mb = (lane >> 4) * 8;
    const int row0 = wm * 32 + mb;            // + sm*16
    const int col0 = wn * 64 + n0;            // + sn*16
    constexpr int RS = (MODE == 0) ? 32768 : 256;   // row stride in floats
    float* base = (MODE == 0)
        ? E + (((size_t)b * 128) * 128 + idx) * 256
        : E + (((size_t)b * 128 + idx) * 128) * 256 + 128;
#pragma unroll
    for (int sm = 0; sm < 2; ++sm)
#pragma unroll
        for (int sn = 0; sn < 4; ++sn)
            store8<RS>(base + (size_t)(row0 + sm * 16) * RS + col0 + sn * 16,
                       acc[sm][sn]);
}

// ---------------------------------------------------------------------------
// Kernel 3: softmax stats over axis=1 (i). For each (b,j,m): max_i, sum_i exp.
// ---------------------------------------------------------------------------
__global__ __launch_bounds__(256) void cc_softmax_stats(
    const float* __restrict__ E, float* __restrict__ Mx, float* __restrict__ Sx)
{
    const int b = blockIdx.x >> 7, j = blockIdx.x & 127;
    const int m = threadIdx.x;                   // 0..255
    const size_t stride = 128 * 256;
    size_t base = (((size_t)b * 128) * 128 + j) * 256 + m;   // i = 0
    float mx = -3.402823e38f;
    for (int i = 0; i < 128; ++i) mx = fmaxf(mx, E[base + (size_t)i * stride]);
    float s = 0.f;
    for (int i = 0; i < 128; ++i) s += __expf(E[base + (size_t)i * stride] - mx);
    size_t o = ((size_t)b * 128 + j) * 256 + m;
    Mx[o] = mx; Sx[o] = s;
}

// ---------------------------------------------------------------------------
// Kernel 4: aggregation (MODE templated, same rationale as cc_energy).
//  MODE==0 (per column j=idx): Y  = attH * V_col (write)
//  MODE==1 (per row    i=idx): Y += attW * V_row (accumulate)
// att recomputed on the fly as exp(E-M)/S, cast f16 into LDS A-fragments.
// Out tile 128 x 512 processed in 8 chunks of 64 cols; K=128.
// ---------------------------------------------------------------------------
template <int MODE>
__global__ __launch_bounds__(256) void cc_aggregate(
    const float* __restrict__ E, const float* __restrict__ Mx,
    const float* __restrict__ Sx, const _Float16* __restrict__ V,
    float* __restrict__ Y)
{
    __shared__ __align__(16) _Float16 Att[128][136];
    __shared__ __align__(16) _Float16 Vt[64][136];
    const int tid = threadIdx.x;
    const int b   = blockIdx.x >> 7;
    const int idx = blockIdx.x & 127;
    const size_t bOff = (size_t)b * 16384;

    // stage attention 128x128 as f16; E/Mx/Sx contiguous in k -> float4.
    // thread's k4 is fixed across iterations; for MODE 0 the stats are too.
    {
        const int k4 = (tid & 31) * 4;
        if (MODE == 0) {
            const size_t sidx = ((size_t)b * 128 + idx) * 256 + k4;
            const float4 mv = *(const float4*)(Mx + sidx);
            const float4 sv = *(const float4*)(Sx + sidx);
#pragma unroll
            for (int it = 0; it < 16; ++it) {
                int r = (tid >> 5) + it * 8;
                float4 ev = *(const float4*)(E + (((size_t)b * 128 + r) * 128 + idx) * 256 + k4);
                v4h h;
                h[0] = (_Float16)(__expf(ev.x - mv.x) / sv.x);
                h[1] = (_Float16)(__expf(ev.y - mv.y) / sv.y);
                h[2] = (_Float16)(__expf(ev.z - mv.z) / sv.z);
                h[3] = (_Float16)(__expf(ev.w - mv.w) / sv.w);
                *(v4h*)&Att[r][k4] = h;
            }
        } else {
#pragma unroll
            for (int it = 0; it < 16; ++it) {
                int r = (tid >> 5) + it * 8;
                size_t eidx = (((size_t)b * 128 + idx) * 128 + r) * 256 + 128 + k4;
                size_t sidx = ((size_t)b * 128 + r) * 256 + 128 + k4;
                float4 ev = *(const float4*)(E + eidx);
                float4 mv = *(const float4*)(Mx + sidx);
                float4 sv = *(const float4*)(Sx + sidx);
                v4h h;
                h[0] = (_Float16)(__expf(ev.x - mv.x) / sv.x);
                h[1] = (_Float16)(__expf(ev.y - mv.y) / sv.y);
                h[2] = (_Float16)(__expf(ev.z - mv.z) / sv.z);
                h[3] = (_Float16)(__expf(ev.w - mv.w) / sv.w);
                *(v4h*)&Att[r][k4] = h;
            }
        }
    }

    const int lane = tid & 31, wave = tid >> 5;
    const int wm = wave >> 1;  // 0..3 : 32 out rows
    const int wn = wave & 1;   // 0..1 : 32 out cols within 64-col chunk

    for (int cc = 0; cc < 8; ++cc) {
        __syncthreads();
        // stage V^T : Vt[c][k] = V[n(k)][cc*64+c]; batch the 4 v8h loads,
        // then scalar LDS scatter for the transpose.
        {
            v8h vv[4];
            int kk[4];
#pragma unroll
            for (int it = 0; it < 4; ++it) {
                int e = tid + it * 256;                 // 1024 chunks of 8 halves
                int k = e >> 3, c8 = (e & 7) * 8;
                int n = (MODE == 0) ? (k * 128 + idx) : (idx * 128 + k);
                vv[it] = *(const v8h*)(V + (bOff + n) * 512 + cc * 64 + c8);
                kk[it] = k;
            }
#pragma unroll
            for (int it = 0; it < 4; ++it) {
                int e = tid + it * 256;
                int c8 = (e & 7) * 8;
#pragma unroll
                for (int t = 0; t < 8; ++t) Vt[c8 + t][kk[it]] = vv[it][t];
            }
        }
        __syncthreads();

        v8f acc[2][2] = {};
#pragma unroll
        for (int k0 = 0; k0 < 128; k0 += 32) {
            v16h af0 = load_frag(&Att[wm * 32][k0], 136);
            v16h af1 = load_frag(&Att[wm * 32 + 16][k0], 136);
#pragma unroll
            for (int sn = 0; sn < 2; ++sn) {
                v16h bf = load_frag(&Vt[wn * 32 + sn * 16][k0], 136);
                acc[0][sn] = wmma_f32_16x16x32(af0, bf, acc[0][sn]);
                acc[1][sn] = wmma_f32_16x16x32(af1, bf, acc[1][sn]);
            }
        }

        // epilogue: constant per-row stride per mode -> immediate offsets
        const int n0 = lane & 15, mb = (lane >> 4) * 8;
        const int row0 = wm * 32 + mb;
        const int col0 = cc * 64 + wn * 32 + n0;
        constexpr int RS = (MODE == 0) ? 65536 : 512;   // row stride in floats
        float* base = (MODE == 0)
            ? Y + (bOff + idx) * 512
            : Y + (bOff + (size_t)idx * 128) * 512;
#pragma unroll
        for (int sm = 0; sm < 2; ++sm)
#pragma unroll
            for (int sn = 0; sn < 2; ++sn) {
                float* p = base + (size_t)(row0 + sm * 16) * RS + col0 + sn * 16;
                if (MODE == 0) store8<RS>(p, acc[sm][sn]);
                else           accum8<RS>(p, acc[sm][sn]);
            }
    }
}

// ---------------------------------------------------------------------------
// Kernel 5: block-1 combine, fused 7x7+5x5+3x3 depthwise convs (L2-resident):
//   feat2[b,n,c] = gamma*Y + x + dwconv(x)
// ---------------------------------------------------------------------------
__global__ __launch_bounds__(256) void combine1(
    const float* __restrict__ x, const float* __restrict__ Y,
    const float* __restrict__ gamma,
    const float* __restrict__ wp7, const float* __restrict__ bp7,
    const float* __restrict__ wp5, const float* __restrict__ bp5,
    const float* __restrict__ wp3, const float* __restrict__ bp3,
    float* __restrict__ feat2)
{
    size_t gid = (size_t)blockIdx.x * 256 + threadIdx.x;  // over B*N*C
    int c = (int)(gid & 511);
    size_t bn = gid >> 9;
    int n = (int)(bn & 16383);
    int b = (int)(bn >> 14);
    int i = n >> 7, j = n & 127;

    float acc = bp7[c] + bp5[c] + bp3[c];
#pragma unroll
    for (int dy = -3; dy <= 3; ++dy) {
        int ii = i + dy; if (ii < 0 || ii > 127) continue;
#pragma unroll
        for (int dx = -3; dx <= 3; ++dx) {
            int jj = j + dx; if (jj < 0 || jj > 127) continue;
            float fv = x[(((size_t)b * 16384) + (size_t)ii * 128 + jj) * 512 + c];
            acc += fv * wp7[c * 49 + (dy + 3) * 7 + (dx + 3)];
            if (dy >= -2 && dy <= 2 && dx >= -2 && dx <= 2)
                acc += fv * wp5[c * 25 + (dy + 2) * 5 + (dx + 2)];
            if (dy >= -1 && dy <= 1 && dx >= -1 && dx <= 1)
                acc += fv * wp3[c * 9 + (dy + 1) * 3 + (dx + 1)];
        }
    }
    feat2[gid] = gamma[0] * Y[gid] + x[gid] + acc;
}

// Kernel 6: block-2 combine -> final output (already (b, n, c) layout).
__global__ __launch_bounds__(256) void combine2(
    const float* __restrict__ feat2, const float* __restrict__ Y,
    const float* __restrict__ gamma1, float* __restrict__ out)
{
    size_t gid = (size_t)blockIdx.x * 256 + threadIdx.x;
    out[gid] = gamma1[0] * Y[gid] + feat2[gid];
}

// ---------------------------------------------------------------------------
extern "C" void kernel_launch(void* const* d_in, const int* in_sizes, int n_in,
                              void* d_out, int out_size, void* d_ws, size_t ws_size,
                              hipStream_t stream)
{
    (void)in_sizes; (void)n_in; (void)out_size; (void)ws_size;
    const float* x     = (const float*)d_in[0];
    const float* wq    = (const float*)d_in[1];
    const float* bq    = (const float*)d_in[2];
    const float* wk    = (const float*)d_in[3];
    const float* bk    = (const float*)d_in[4];
    const float* wv    = (const float*)d_in[5];
    const float* bv    = (const float*)d_in[6];
    const float* gamma = (const float*)d_in[7];
    const float* wq1   = (const float*)d_in[8];
    const float* bq1   = (const float*)d_in[9];
    const float* wk1   = (const float*)d_in[10];
    const float* bk1   = (const float*)d_in[11];
    const float* wv1   = (const float*)d_in[12];
    const float* bv1   = (const float*)d_in[13];
    const float* gamma1= (const float*)d_in[14];
    const float* wp7   = (const float*)d_in[15];
    const float* bp7   = (const float*)d_in[16];
    const float* wp5   = (const float*)d_in[17];
    const float* bp5   = (const float*)d_in[18];
    const float* wp3   = (const float*)d_in[19];
    const float* bp3   = (const float*)d_in[20];
    float* out = (float*)d_out;

    const int M = 32768, C = 512, Cq = 64;

    char* ws = (char*)d_ws;
    _Float16* qh = (_Float16*)ws;  ws += (size_t)M * Cq * 2;
    _Float16* kh = (_Float16*)ws;  ws += (size_t)M * Cq * 2;
    _Float16* vh = (_Float16*)ws;  ws += (size_t)M * C * 2;
    float* E     = (float*)ws;     ws += (size_t)2 * 128 * 128 * 256 * 4;
    float* Mx    = (float*)ws;     ws += (size_t)2 * 128 * 256 * 4;
    float* Sx    = (float*)ws;     ws += (size_t)2 * 128 * 256 * 4;
    float* Y     = (float*)ws;     ws += (size_t)M * C * 4;
    float* feat2 = (float*)ws;     ws += (size_t)M * C * 4;

    dim3 blk(256);

    // ---- CC block 1 ----
    gemm512<64> <<<dim3(M / 64, 1), blk, 0, stream>>>(x, wq, bq, qh);
    gemm512<64> <<<dim3(M / 64, 1), blk, 0, stream>>>(x, wk, bk, kh);
    gemm512<512><<<dim3(M / 64, 8), blk, 0, stream>>>(x, wv, bv, vh);
    cc_energy<0><<<dim3(256), blk, 0, stream>>>(qh, kh, E);
    cc_energy<1><<<dim3(256), blk, 0, stream>>>(qh, kh, E);
    cc_softmax_stats<<<dim3(256), blk, 0, stream>>>(E, Mx, Sx);
    cc_aggregate<0><<<dim3(256), blk, 0, stream>>>(E, Mx, Sx, vh, Y);
    cc_aggregate<1><<<dim3(256), blk, 0, stream>>>(E, Mx, Sx, vh, Y);
    combine1<<<dim3(65536), blk, 0, stream>>>(x, Y, gamma, wp7, bp7, wp5, bp5, wp3, bp3, feat2);

    // ---- CC block 2 ----
    gemm512<64> <<<dim3(M / 64, 1), blk, 0, stream>>>(feat2, wq1, bq1, qh);
    gemm512<64> <<<dim3(M / 64, 1), blk, 0, stream>>>(feat2, wk1, bk1, kh);
    gemm512<512><<<dim3(M / 64, 8), blk, 0, stream>>>(feat2, wv1, bv1, vh);
    cc_energy<0><<<dim3(256), blk, 0, stream>>>(qh, kh, E);
    cc_energy<1><<<dim3(256), blk, 0, stream>>>(qh, kh, E);
    cc_softmax_stats<<<dim3(256), blk, 0, stream>>>(E, Mx, Sx);
    cc_aggregate<0><<<dim3(256), blk, 0, stream>>>(E, Mx, Sx, vh, Y);
    cc_aggregate<1><<<dim3(256), blk, 0, stream>>>(E, Mx, Sx, vh, Y);
    combine2<<<dim3(65536), blk, 0, stream>>>(feat2, Y, gamma1, out);
}
